// MimiResidualVectorQuantizer_45466523795679
// MI455X (gfx1250) — compile-verified, matrix operand order checked by reading, not run on hardware
//
#include <hip/hip_runtime.h>
#include <hip/hip_bf16.h>

// ---- types for WMMA / TDM operands -----------------------------------------
typedef __attribute__((ext_vector_type(16))) __bf16        bf16x16;
typedef __attribute__((ext_vector_type(8)))  float         f32x8;
typedef __attribute__((ext_vector_type(4)))  float         f32x4;
typedef __attribute__((ext_vector_type(4)))  unsigned int  u32x4;
typedef __attribute__((ext_vector_type(8)))  int           i32x8;
typedef __attribute__((ext_vector_type(4)))  int           i32x4;

union BV { bf16x16 v; u32x4 q[2]; };   // 16 bf16 = 32B = two 16B chunks
union FV { f32x8  v; f32x4 f4[2]; };

__device__ __forceinline__ unsigned short f2bf(float x) {
  unsigned int u = __float_as_uint(x);
  unsigned int r = u + 0x7FFFu + ((u >> 16) & 1u);   // round-to-nearest-even
  return (unsigned short)(r >> 16);
}

// Problem constants
#define BATCH 8
#define CIN   512
#define TLEN  4096
#define DDIM  256
#define KQ    8
#define NCB   2048
#define MROWS (BATCH*TLEN)           // 32768
#define CODES_ELEMS (KQ*BATCH*TLEN)  // 262144

// ---- 1) codebook prep: e = sum/clamp(usage); f32 + bf16 + ||e||^2 -----------
__global__ __launch_bounds__(256) void k_prep_codebook(
    const float* __restrict__ embed_sum, const float* __restrict__ usage,
    float* __restrict__ ef32, unsigned short* __restrict__ ebf,
    float* __restrict__ e2) {
  const int kn = blockIdx.x;          // 0..K*N-1
  const int d  = threadIdx.x;         // 0..255
  float u = usage[kn];
  u = (u < 1e-5f) ? 1e-5f : u;
  float e = embed_sum[(size_t)kn * DDIM + d] / u;
  ef32[(size_t)kn * DDIM + d] = e;
  ebf [(size_t)kn * DDIM + d] = f2bf(e);
  __shared__ float red[256];
  red[d] = e * e;
  __syncthreads();
  for (int s = 128; s > 0; s >>= 1) {
    if (d < s) red[d] += red[d + s];
    __syncthreads();
  }
  if (d == 0) e2[kn] = red[0];
}

// ---- 2) weight bf16 conversion ---------------------------------------------
__global__ __launch_bounds__(256) void k_prep_weights(
    const float* __restrict__ w_in, const float* __restrict__ w_out,
    unsigned short* __restrict__ wibf, unsigned short* __restrict__ wobf) {
  int i = blockIdx.x * 256 + threadIdx.x;   // grid 512 -> 131072 threads
  wibf[i] = f2bf(w_in[i]);
  wobf[i] = f2bf(w_out[i]);
}

// ---- 3) input projection: x[r,d] = sum_c w_in[d,c]*emb[b,c,t] ---------------
// Writes X0, initial residual R (f32) and its bf16 mirror Rbf.
__global__ __launch_bounds__(256) void k_proj_in(
    const float* __restrict__ emb, const unsigned short* __restrict__ wibf,
    float* __restrict__ X0, float* __restrict__ R,
    unsigned short* __restrict__ Rbf) {
  __shared__ __align__(16) unsigned short As[16 * CIN];   // 16 rows x K=512 bf16
  const int tid = threadIdx.x;
  const int r0  = blockIdx.x * 16;
  const int b   = r0 >> 12;             // /TLEN
  const int t0  = r0 & (TLEN - 1);
  const float* eb = emb + (size_t)b * CIN * TLEN;
  for (int idx = tid; idx < 16 * CIN; idx += 256) {
    int m = idx & 15, kk = idx >> 4;
    As[m * CIN + kk] = f2bf(eb[(size_t)kk * TLEN + t0 + m]);
  }
  __syncthreads();
  const int wave = tid >> 5, lane = tid & 31;
  const int l16 = lane & 15, half = lane >> 4;
  for (int it = 0; it < 2; ++it) {
    const int n0 = wave * 16 + it * 128;
    f32x8 acc = {};
    for (int k0 = 0; k0 < CIN; k0 += 32) {
      BV a, bb;
      const unsigned short* ap = &As[l16 * CIN + k0 + half * 8];
      a.q[0] = *(const u32x4*)(ap);
      a.q[1] = *(const u32x4*)(ap + 16);
      const unsigned short* bp = wibf + (size_t)(n0 + l16) * CIN + k0 + half * 8;
      bb.q[0] = *(const u32x4*)(bp);
      bb.q[1] = *(const u32x4*)(bp + 16);
      acc = __builtin_amdgcn_wmma_f32_16x16x32_bf16(
          false, a.v, false, bb.v, (short)0, acc, false, false);
    }
    const int n = n0 + l16;
    for (int i = 0; i < 8; ++i) {
      int m = i + half * 8;
      size_t off = (size_t)(r0 + m) * DDIM + n;
      X0[off]  = acc[i];
      R[off]   = acc[i];
      Rbf[off] = f2bf(acc[i]);
    }
  }
}

// ---- 4) fused distance + argmin + residual update (one quantizer) -----------
// M-tile = 64 rows/block: 4 WMMA accumulator chains per wave reuse each B-tile
// 4x (quarters the codebook L2 traffic). A-tile is DMA'd to LDS by the TDM.
__global__ __launch_bounds__(256) void k_dist(
    float* __restrict__ R, unsigned short* __restrict__ Rbf,
    const unsigned short* __restrict__ cb_bf,
    const float* __restrict__ cb_f32, const float* __restrict__ e2,
    float* __restrict__ codes_out, int kq) {
  // 64 KB shared: phase 1 uses first 32 KB as the bf16 A-tile (64x256);
  // phase 2 (after a barrier) reuses all 64 KB for the argmin reduction.
  __shared__ __align__(16) unsigned char smem[65536];
  __shared__ int sIdx[64];
  unsigned short* As = (unsigned short*)smem;          // 64*256 bf16 = 32 KB
  float* rdm = (float*)smem;                           // 64*128 f32  = 32 KB
  int*   rim = (int*)(smem + 32768);                   // 64*128 i32  = 32 KB

  const int tid = threadIdx.x;
  const int r0  = blockIdx.x * 64;

  // ---- TDM: one contiguous 32 KB copy  Rbf[r0*256 ..] -> LDS As -------------
  if (tid < 32) {  // wave 0 only (TDM ignores EXEC; issue once)
    unsigned long long ga = (unsigned long long)(size_t)(Rbf + (size_t)r0 * DDIM);
    unsigned int la = (unsigned int)(size_t)As;        // low 32 bits = LDS offset
    // D# group0: count=1 | lds_addr | global_addr[56:0] | type=2
    u32x4 g0 = { 1u, la,
                 (unsigned int)(ga & 0xFFFFFFFFu),
                 (unsigned int)((ga >> 32) & 0x01FFFFFFu) | (2u << 30) };
    // D# group1: data_size=2B; tensor_dim0=16384; tensor_dim1=1;
    //            tile_dim0=16384 (1-D); stride0=16384
    i32x8 g1 = { (int)(1u << 16),
                 (int)(0x4000u << 16),
                 (int)(1u << 16),
                 (int)(0x4000u << 16),
                 0,
                 16384,
                 0, 0 };
    i32x4 gz  = { 0, 0, 0, 0 };
    i32x8 gz8 = { 0, 0, 0, 0, 0, 0, 0, 0 };
    __builtin_amdgcn_tensor_load_to_lds(g0, g1, gz, gz, gz8, 0);
    __builtin_amdgcn_s_wait_tensorcnt(0);
  }
  __syncthreads();

  const int wave = tid >> 5, lane = tid & 31;
  const int l16 = lane & 15, half = lane >> 4;
  const unsigned short* cbk = cb_bf + (size_t)kq * NCB * DDIM;
  const float* e2k = e2 + kq * NCB;

  float bestd[4][8];
  int   bestn[4][8];
#pragma unroll
  for (int mt = 0; mt < 4; ++mt)
#pragma unroll
    for (int i = 0; i < 8; ++i) { bestd[mt][i] = 3.4e38f; bestn[mt][i] = 0; }

  const unsigned short* arow = As + (size_t)l16 * DDIM + half * 8;

  for (int nt = 0; nt < 16; ++nt) {
    const int n = wave * 16 + nt * 128 + l16;
    if (nt < 15)  // speculative prefetch of next codebook tile
      __builtin_prefetch(cbk + (size_t)(n + 128) * DDIM, 0, 1);
    f32x8 acc[4] = {};
    const unsigned short* brow = cbk + (size_t)n * DDIM + half * 8;
#pragma unroll
    for (int kk = 0; kk < 8; ++kk) {
      BV bb;
      bb.q[0] = *(const u32x4*)(brow + kk * 32);
      bb.q[1] = *(const u32x4*)(brow + kk * 32 + 16);
#pragma unroll
      for (int mt = 0; mt < 4; ++mt) {
        BV a;
        const unsigned short* ap = arow + mt * 16 * DDIM + kk * 32;
        a.q[0] = *(const u32x4*)(ap);
        a.q[1] = *(const u32x4*)(ap + 16);
        acc[mt] = __builtin_amdgcn_wmma_f32_16x16x32_bf16(
            false, a.v, false, bb.v, (short)0, acc[mt], false, false);
      }
    }
    const float e2n = e2k[n];
#pragma unroll
    for (int mt = 0; mt < 4; ++mt)
#pragma unroll
      for (int i = 0; i < 8; ++i) {
        float d = e2n - 2.0f * acc[mt][i];
        if (d < bestd[mt][i]) { bestd[mt][i] = d; bestn[mt][i] = n; }
      }
  }

  __syncthreads();   // all waves done reading As -> safe to reuse as rdm/rim
#pragma unroll
  for (int mt = 0; mt < 4; ++mt)
#pragma unroll
    for (int i = 0; i < 8; ++i) {
      int row  = mt * 16 + half * 8 + i;        // 0..63
      int slot = row * 128 + wave * 16 + l16;
      rdm[slot] = bestd[mt][i];
      rim[slot] = bestn[mt][i];
    }
  __syncthreads();

  if (tid < 64) {
    float bd = 3.4e38f; int bn = 0x7fffffff;
    for (int s = 0; s < 128; ++s) {
      float d = rdm[tid * 128 + s];
      int   n = rim[tid * 128 + s];
      if (d < bd || (d == bd && n < bn)) { bd = d; bn = n; }  // first-occurrence
    }
    sIdx[tid] = bn;
    codes_out[(size_t)kq * MROWS + r0 + tid] = (float)bn;
  }
  __syncthreads();

  // residual update: r -= e[argmin]; refresh bf16 mirror (4 threads x 64 f/row)
  {
    int row = tid >> 2, j = tid & 3;
    int n = sIdx[row];
    const float* e = cb_f32 + ((size_t)kq * NCB + n) * DDIM + j * 64;
    float* rr = R + (size_t)(r0 + row) * DDIM + j * 64;
    unsigned short* rb = Rbf + (size_t)(r0 + row) * DDIM + j * 64;
    for (int dd = 0; dd < 64; ++dd) {
      float v = rr[dd] - e[dd];
      rr[dd] = v;
      rb[dd] = f2bf(v);
    }
  }
}

// ---- 5) output projection: out[b,c,t] = sum_d w_out[c,d]*(X0-R)[r,d] --------
__global__ __launch_bounds__(256) void k_proj_out(
    const float* __restrict__ X0, const float* __restrict__ R,
    const unsigned short* __restrict__ wobf, float* __restrict__ out) {
  __shared__ __align__(16) unsigned short As[16 * DDIM];
  const int tid = threadIdx.x;
  const int r0  = blockIdx.x * 16;
  for (int idx = tid; idx < 16 * DDIM; idx += 256) {
    int m = idx >> 8, kk = idx & 255;
    size_t off = (size_t)(r0 + m) * DDIM + kk;
    As[m * DDIM + kk] = f2bf(X0[off] - R[off]);
  }
  __syncthreads();
  const int wave = tid >> 5, lane = tid & 31;
  const int l16 = lane & 15, half = lane >> 4;
  BV a[8];
#pragma unroll
  for (int kk = 0; kk < 8; ++kk) {
    const unsigned short* ap = &As[l16 * DDIM + kk * 32 + half * 8];
    a[kk].q[0] = *(const u32x4*)(ap);
    a[kk].q[1] = *(const u32x4*)(ap + 16);
  }
  const int b  = r0 >> 12;
  const int t0 = r0 & (TLEN - 1);
  for (int it = 0; it < 4; ++it) {
    const int c = wave * 16 + it * 128 + l16;
    f32x8 acc = {};
    const unsigned short* brow = wobf + (size_t)c * DDIM + half * 8;
#pragma unroll
    for (int kk = 0; kk < 8; ++kk) {
      BV bb;
      bb.q[0] = *(const u32x4*)(brow + kk * 32);
      bb.q[1] = *(const u32x4*)(brow + kk * 32 + 16);
      acc = __builtin_amdgcn_wmma_f32_16x16x32_bf16(
          false, a[kk].v, false, bb.v, (short)0, acc, false, false);
    }
    FV f; f.v = acc;
    // rows m=i+8*half map to consecutive t -> two 16B stores per lane
    float* op = out + (size_t)b * CIN * TLEN + (size_t)c * TLEN + t0 + half * 8;
    *(f32x4*)(op)     = f.f4[0];
    *(f32x4*)(op + 4) = f.f4[1];
  }
}

// ---- host launcher ----------------------------------------------------------
extern "C" void kernel_launch(void* const* d_in, const int* in_sizes, int n_in,
                              void* d_out, int out_size, void* d_ws, size_t ws_size,
                              hipStream_t stream) {
  const float* embeddings = (const float*)d_in[0];  // [8,512,4096]
  const float* w_in       = (const float*)d_in[1];  // [256,512]
  const float* w_out      = (const float*)d_in[2];  // [512,256]
  const float* embed_sum  = (const float*)d_in[3];  // [8,2048,256]
  const float* usage      = (const float*)d_in[4];  // [8,2048]
  float* out = (float*)d_out;                       // codes (K*B*T) then out (B*512*T)

  char* p = (char*)d_ws;
  float* X0   = (float*)p; p += (size_t)MROWS * DDIM * 4;      // 32 MB
  float* R    = (float*)p; p += (size_t)MROWS * DDIM * 4;      // 32 MB
  unsigned short* Rbf = (unsigned short*)p; p += (size_t)MROWS * DDIM * 2; // 16 MB
  float* ef32 = (float*)p; p += (size_t)KQ * NCB * DDIM * 4;   // 16 MB
  float* e2   = (float*)p; p += (size_t)KQ * NCB * 4;          // 64 KB
  unsigned short* ebf  = (unsigned short*)p; p += (size_t)KQ * NCB * DDIM * 2; // 8 MB
  unsigned short* wibf = (unsigned short*)p; p += (size_t)DDIM * CIN * 2;
  unsigned short* wobf = (unsigned short*)p; p += (size_t)CIN * DDIM * 2;

  k_prep_codebook<<<KQ * NCB, 256, 0, stream>>>(embed_sum, usage, ef32, ebf, e2);
  k_prep_weights<<<512, 256, 0, stream>>>(w_in, w_out, wibf, wobf);
  k_proj_in<<<MROWS / 16, 256, 0, stream>>>(embeddings, wibf, X0, R, Rbf);
  for (int k = 0; k < KQ; ++k)
    k_dist<<<MROWS / 64, 256, 0, stream>>>(R, Rbf, ebf, ef32, e2, out, k);
  k_proj_out<<<MROWS / 16, 256, 0, stream>>>(X0, R, wobf, out + CODES_ELEMS);
}
